// PuzzleCNN_25623774888453
// MI455X (gfx1250) — compile-verified, matrix operand order checked by reading, not run on hardware
//
#include <hip/hip_runtime.h>
#include <hip/hip_bf16.h>
#include <stdint.h>

typedef __bf16 bf16_t;
typedef __attribute__((ext_vector_type(16))) __bf16 v16bf;
typedef __attribute__((ext_vector_type(8)))  float  v8f;

#define NBATCH 4096
#define CHUNK  1024

union FragBF { v16bf v; uint32_t u[8]; };

// -------------------------------------------------------------------------
// Patch shuffle: argsort(noise) per image, gather patches, emit channels-last
// bf16 image [N*32*32, 3] and target ids_restore (as float) into d_out tail.
// -------------------------------------------------------------------------
__global__ __launch_bounds__(256)
void shuffle_kernel(const float* __restrict__ x, const float* __restrict__ noise,
                    bf16_t* __restrict__ img_out, float* __restrict__ target_out)
{
    const int n = blockIdx.x;
    const int t = threadIdx.x;
    __shared__ float s_noise[16];
    __shared__ int   ids_shuffle[16];   // ids_shuffle[r] = patch placed at slot r

    if (t < 16) s_noise[t] = noise[n * 16 + t];
    __syncthreads();
    if (t < 16) {
        float v = s_noise[t];
        int rank = 0;
        for (int j = 0; j < 16; ++j) {
            float u = s_noise[j];
            rank += (u < v) || (u == v && j < t);   // stable ascending
        }
        ids_shuffle[rank] = t;
        target_out[n * 16 + t] = (float)rank;       // ids_restore[t] == rank
    }
    __syncthreads();

    // out[n, y, x, c] = in[n, c, py, px]
    for (int idx = t; idx < 32 * 32 * 3; idx += 256) {
        int c   = idx % 3;
        int pix = idx / 3;
        int xx  = pix % 32;
        int yy  = pix / 32;
        int slot = (yy >> 3) * 4 + (xx >> 3);
        int p    = ids_shuffle[slot];
        int py   = (p >> 2) * 8 + (yy & 7);
        int px   = (p & 3)  * 8 + (xx & 7);
        img_out[(size_t)n * 3072 + pix * 3 + c] =
            (bf16_t)x[((size_t)n * 3 + c) * 1024 + py * 32 + px];
    }
}

// -------------------------------------------------------------------------
// Weight conversion (f32 -> bf16). All weights stored N-MAJOR: [Cout][Kpad]
// so GEMM B-tiles are contiguous along k (vectorizable staging).
// -------------------------------------------------------------------------
__global__ __launch_bounds__(256)
void convert_conv_w(const float* __restrict__ w, bf16_t* __restrict__ out,
                    int Cin, int Cout, int Kpad)
{
    int idx = blockIdx.x * 256 + threadIdx.x;
    if (idx >= Kpad * Cout) return;
    int k = idx % Kpad, o = idx / Kpad;
    float v = 0.f;
    int K = Cin * 9;
    if (k < K) {
        int cin = k % Cin;
        int s   = k / Cin;             // ky*3+kx
        v = w[(size_t)o * K + cin * 9 + s];
    }
    out[idx] = (bf16_t)v;              // out[o][k]
}

// fc weights: reference [K][N] -> transposed bf16 [N][K]
__global__ __launch_bounds__(256)
void convert_fc_w(const float* __restrict__ w, bf16_t* __restrict__ out,
                  int K, int N)
{
    int idx = blockIdx.x * 256 + threadIdx.x;   // over K*N, out flat = n*K+k
    if (idx >= K * N) return;
    int k = idx % K, n = idx / K;
    out[idx] = (bf16_t)w[(size_t)k * N + n];
}

// FC1: our feature order is (y*4+x)*128 + c; reference order is c*16 + y*4 + x
__global__ __launch_bounds__(256)
void convert_fc1_w(const float* __restrict__ w, bf16_t* __restrict__ out)
{
    int idx = blockIdx.x * 256 + threadIdx.x;   // over 2048*1024, out flat = o*2048+k
    int k = idx & 2047, o = idx >> 11;
    int c = k & 127, s = k >> 7;
    out[idx] = (bf16_t)w[(size_t)(c * 16 + s) * 1024 + o];
}

// -------------------------------------------------------------------------
// Implicit-im2col conv GEMM + bias + ReLU + fused 2x2 maxpool.
// in  : [Mc, CIN] channels-last bf16 (Mc = images*H*W)
// wB  : [COUT][Kpad] bf16 (n-major),  bias: [COUT] f32
// out : pooled [Mc/4, COUT] bf16
// block = 256 threads = 8 waves in a 4x2 wave grid; BM=128.
// -------------------------------------------------------------------------
template<int CIN, int COUT, int H, int W, int BN>
__global__ __launch_bounds__(256)
void conv_pool_gemm(const bf16_t* __restrict__ in, const bf16_t* __restrict__ wB,
                    const float* __restrict__ bias, bf16_t* __restrict__ out)
{
    constexpr int BM  = 128;
    constexpr int K   = CIN * 9;
    constexpr int KP  = (K + 31) & ~31;
    constexpr int LDA = 36;            // 32 + 4 pad: 8B-aligned groups, no bank conflicts
    constexpr int NT  = BN / 32;       // wmma n-tiles per wave
    constexpr int Wp  = W / 2, Hp = H / 2;

    __shared__ bf16_t Alds[BM * LDA];
    __shared__ bf16_t Blds[BN * LDA];
    __shared__ float  Clds[BM * (BN + 1)];

    const int tid  = threadIdx.x;
    const int lane = tid & 31;
    const int l16  = lane & 15;
    const int hf   = lane >> 4;
    const int wm   = (tid >> 5) >> 1;  // 0..3
    const int wn   = (tid >> 5) & 1;   // 0..1
    const int m0   = blockIdx.x * BM;
    const int n0   = blockIdx.y * BN;

    v8f acc[2][NT];
    v8f zero = {};
    #pragma unroll
    for (int mt = 0; mt < 2; ++mt)
        #pragma unroll
        for (int nt = 0; nt < NT; ++nt) acc[mt][nt] = zero;

    for (int k0 = 0; k0 < KP; k0 += 32) {
        // --- stage A tile (on-the-fly im2col) ---
        if constexpr (CIN % 4 == 0) {
            // 4 consecutive k = 4 consecutive cin inside one (ky,kx) tap:
            // contiguous in channels-last memory -> 8B vector loads.
            #pragma unroll
            for (int i = 0; i < (BM * 8) / 256; ++i) {
                int idx = tid + i * 256;
                int r = idx >> 3, j = idx & 7;
                int k = k0 + 4 * j;
                int cin = k % CIN;
                int s   = k / CIN;
                int m   = m0 + r;
                int xx  = (m % W) + (s % 3) - 1;
                int tt  = m / W;
                int yy  = (tt % H) + (s / 3) - 1;
                int ni  = tt / H;
                uint64_t val = 0;
                if ((unsigned)yy < (unsigned)H && (unsigned)xx < (unsigned)W)
                    val = *(const uint64_t*)&in[((size_t)ni * (H * W) + yy * W + xx) * CIN + cin];
                *(uint64_t*)&Alds[r * LDA + 4 * j] = val;
            }
        } else {
            for (int idx = tid; idx < BM * 32; idx += 256) {
                int r = idx >> 5, kk = idx & 31, k = k0 + kk;
                float v = 0.f;
                if (k < K) {
                    int m   = m0 + r;
                    int cin = k % CIN;
                    int s   = k / CIN;
                    int xx  = (m % W) + (s % 3) - 1;
                    int tt  = m / W;
                    int yy  = (tt % H) + (s / 3) - 1;
                    int ni  = tt / H;
                    if ((unsigned)yy < (unsigned)H && (unsigned)xx < (unsigned)W)
                        v = (float)in[((size_t)ni * (H * W) + yy * W + xx) * CIN + cin];
                }
                Alds[r * LDA + kk] = (bf16_t)v;
            }
        }
        // --- stage B tile: wB is n-major, rows contiguous in k -> 8B loads ---
        #pragma unroll
        for (int i = 0; i < (BN * 8) / 256; ++i) {
            int idx = tid + i * 256;
            int nn = idx >> 3, j = idx & 7;
            *(uint64_t*)&Blds[nn * LDA + 4 * j] =
                *(const uint64_t*)&wB[(size_t)(n0 + nn) * KP + k0 + 4 * j];
        }
        __syncthreads();

        FragBF a[2], b[NT];
        #pragma unroll
        for (int mt = 0; mt < 2; ++mt) {
            int row = wm * 32 + mt * 16 + l16;
            #pragma unroll
            for (int vv = 0; vv < 8; ++vv) {
                int kb = 2 * (vv & 3) + ((vv >> 2) << 4) + (hf << 3);
                a[mt].u[vv] = *(const uint32_t*)&Alds[row * LDA + kb];
            }
        }
        #pragma unroll
        for (int nt = 0; nt < NT; ++nt) {
            int col = wn * (BN / 2) + nt * 16 + l16;
            #pragma unroll
            for (int vv = 0; vv < 8; ++vv) {
                int kb = 2 * (vv & 3) + ((vv >> 2) << 4) + (hf << 3);
                b[nt].u[vv] = *(const uint32_t*)&Blds[col * LDA + kb];
            }
        }
        #pragma unroll
        for (int mt = 0; mt < 2; ++mt)
            #pragma unroll
            for (int nt = 0; nt < NT; ++nt)
                acc[mt][nt] = __builtin_amdgcn_wmma_f32_16x16x32_bf16(
                    false, a[mt].v, false, b[nt].v, (short)0, acc[mt][nt], false, false);
        __syncthreads();
    }

    // --- bias + ReLU into LDS, then fused 2x2 maxpool (tile-local) ---
    #pragma unroll
    for (int mt = 0; mt < 2; ++mt)
        #pragma unroll
        for (int nt = 0; nt < NT; ++nt) {
            int col = wn * (BN / 2) + nt * 16 + l16;
            float bv = bias[n0 + col];
            #pragma unroll
            for (int vv = 0; vv < 8; ++vv) {
                int row = wm * 32 + mt * 16 + vv + 8 * hf;
                Clds[row * (BN + 1) + col] = fmaxf(acc[mt][nt][vv] + bv, 0.f);
            }
        }
    __syncthreads();

    const int mp0 = m0 >> 2;
    for (int idx = tid; idx < (BM / 4) * BN; idx += 256) {
        int q = idx / BN, c = idx % BN;
        int mp = mp0 + q;
        int x2 = mp % Wp;
        int tt = mp / Wp;
        int y2 = tt % Hp;
        int ni = tt / Hp;
        int r00 = ni * (H * W) + (2 * y2) * W + 2 * x2 - m0;
        float v0 = Clds[(r00)         * (BN + 1) + c];
        float v1 = Clds[(r00 + 1)     * (BN + 1) + c];
        float v2 = Clds[(r00 + W)     * (BN + 1) + c];
        float v3 = Clds[(r00 + W + 1) * (BN + 1) + c];
        out[(size_t)mp * COUT + n0 + c] = (bf16_t)fmaxf(fmaxf(v0, v1), fmaxf(v2, v3));
    }
}

// -------------------------------------------------------------------------
// Dense FC GEMM: out[M,N] = relu?(A[M,K] @ B^T + bias), Bw n-major [N][K].
// Software-pipelined: next k-tile fetched to registers during WMMA.
// -------------------------------------------------------------------------
template<bool RELU, bool OUTBF>
__global__ __launch_bounds__(256)
void fc_gemm(const bf16_t* __restrict__ A, const bf16_t* __restrict__ Bw,
             const float* __restrict__ bias, void* __restrict__ outp,
             int M, int K, int N)
{
    constexpr int BM = 128, BN = 64, LDA = 36, NT = 2;
    __shared__ bf16_t Alds[BM * LDA];
    __shared__ bf16_t Blds[BN * LDA];

    const int tid  = threadIdx.x;
    const int lane = tid & 31;
    const int l16  = lane & 15;
    const int hf   = lane >> 4;
    const int wm   = (tid >> 5) >> 1;
    const int wn   = (tid >> 5) & 1;
    const int m0   = blockIdx.x * BM;
    const int n0   = blockIdx.y * BN;

    v8f acc[2][NT];
    v8f zero = {};
    #pragma unroll
    for (int mt = 0; mt < 2; ++mt)
        #pragma unroll
        for (int nt = 0; nt < NT; ++nt) acc[mt][nt] = zero;

    uint64_t pa[4], pb[2];
    auto load_tile = [&](int k0) {
        #pragma unroll
        for (int i = 0; i < 4; ++i) {           // BM*8 / 256
            int idx = tid + i * 256;
            int r = idx >> 3, j = idx & 7;
            pa[i] = *(const uint64_t*)&A[(size_t)(m0 + r) * K + k0 + 4 * j];
        }
        #pragma unroll
        for (int i = 0; i < 2; ++i) {           // BN*8 / 256
            int idx = tid + i * 256;
            int nn = idx >> 3, j = idx & 7;
            pb[i] = *(const uint64_t*)&Bw[(size_t)(n0 + nn) * K + k0 + 4 * j];
        }
    };
    auto store_tile = [&]() {
        #pragma unroll
        for (int i = 0; i < 4; ++i) {
            int idx = tid + i * 256;
            int r = idx >> 3, j = idx & 7;
            *(uint64_t*)&Alds[r * LDA + 4 * j] = pa[i];
        }
        #pragma unroll
        for (int i = 0; i < 2; ++i) {
            int idx = tid + i * 256;
            int nn = idx >> 3, j = idx & 7;
            *(uint64_t*)&Blds[nn * LDA + 4 * j] = pb[i];
        }
    };

    load_tile(0);
    for (int k0 = 0; k0 < K; k0 += 32) {
        store_tile();
        __syncthreads();
        if (k0 + 32 < K) load_tile(k0 + 32);    // overlap next fetch with WMMA

        FragBF a[2], b[NT];
        #pragma unroll
        for (int mt = 0; mt < 2; ++mt) {
            int row = wm * 32 + mt * 16 + l16;
            #pragma unroll
            for (int vv = 0; vv < 8; ++vv) {
                int kb = 2 * (vv & 3) + ((vv >> 2) << 4) + (hf << 3);
                a[mt].u[vv] = *(const uint32_t*)&Alds[row * LDA + kb];
            }
        }
        #pragma unroll
        for (int nt = 0; nt < NT; ++nt) {
            int col = wn * (BN / 2) + nt * 16 + l16;
            #pragma unroll
            for (int vv = 0; vv < 8; ++vv) {
                int kb = 2 * (vv & 3) + ((vv >> 2) << 4) + (hf << 3);
                b[nt].u[vv] = *(const uint32_t*)&Blds[col * LDA + kb];
            }
        }
        #pragma unroll
        for (int mt = 0; mt < 2; ++mt)
            #pragma unroll
            for (int nt = 0; nt < NT; ++nt)
                acc[mt][nt] = __builtin_amdgcn_wmma_f32_16x16x32_bf16(
                    false, a[mt].v, false, b[nt].v, (short)0, acc[mt][nt], false, false);
        __syncthreads();
    }

    #pragma unroll
    for (int mt = 0; mt < 2; ++mt)
        #pragma unroll
        for (int nt = 0; nt < NT; ++nt) {
            int colg = n0 + wn * (BN / 2) + nt * 16 + l16;
            float bv = bias[colg];
            #pragma unroll
            for (int vv = 0; vv < 8; ++vv) {
                int rowg = m0 + wm * 32 + mt * 16 + vv + 8 * hf;
                float val = acc[mt][nt][vv] + bv;
                if (RELU) val = fmaxf(val, 0.f);
                if (OUTBF)
                    ((bf16_t*)outp)[(size_t)rowg * N + colg] = (bf16_t)val;
                else
                    ((float*)outp)[(size_t)rowg * N + colg] = val;
            }
        }
}

// -------------------------------------------------------------------------
extern "C" void kernel_launch(void* const* d_in, const int* in_sizes, int n_in,
                              void* d_out, int out_size, void* d_ws, size_t ws_size,
                              hipStream_t stream)
{
    (void)in_sizes; (void)n_in; (void)out_size; (void)ws_size;
    const float* x   = (const float*)d_in[0];
    const float* nz  = (const float*)d_in[1];
    const float* c1w = (const float*)d_in[2];  const float* c1b = (const float*)d_in[3];
    const float* c2w = (const float*)d_in[4];  const float* c2b = (const float*)d_in[5];
    const float* c3w = (const float*)d_in[6];  const float* c3b = (const float*)d_in[7];
    const float* f1w = (const float*)d_in[8];  const float* f1b = (const float*)d_in[9];
    const float* f2w = (const float*)d_in[10]; const float* f2b = (const float*)d_in[11];
    const float* f3w = (const float*)d_in[12]; const float* f3b = (const float*)d_in[13];
    const float* f4w = (const float*)d_in[14]; const float* f4b = (const float*)d_in[15];
    const float* f5w = (const float*)d_in[16]; const float* f5b = (const float*)d_in[17];

    float* out_logits = (float*)d_out;                       // [4096,16,16]
    float* out_target = out_logits + (size_t)NBATCH * 256;   // [4096,16]

    char* p = (char*)d_ws;
    auto alloc = [&](size_t bytes) -> char* {
        char* r = p; p += (bytes + 255) & ~(size_t)255; return r;
    };
    bf16_t* img  = (bf16_t*)alloc((size_t)NBATCH * 1024 * 3 * 2);
    bf16_t* w1   = (bf16_t*)alloc((size_t)32  * 32  * 2);
    bf16_t* w2   = (bf16_t*)alloc((size_t)288 * 64  * 2);
    bf16_t* w3   = (bf16_t*)alloc((size_t)576 * 128 * 2);
    bf16_t* fw1  = (bf16_t*)alloc((size_t)2048 * 1024 * 2);
    bf16_t* fw2  = (bf16_t*)alloc((size_t)1024 * 1024 * 2);
    bf16_t* fw3  = (bf16_t*)alloc((size_t)1024 * 1024 * 2);
    bf16_t* fw4  = (bf16_t*)alloc((size_t)1024 * 1024 * 2);
    bf16_t* fw5  = (bf16_t*)alloc((size_t)1024 * 256  * 2);
    bf16_t* act1 = (bf16_t*)alloc((size_t)CHUNK * 256 * 32  * 2);
    bf16_t* act2 = (bf16_t*)alloc((size_t)CHUNK * 64  * 64  * 2);
    bf16_t* act3 = (bf16_t*)alloc((size_t)NBATCH * 16 * 128 * 2);
    bf16_t* fa   = (bf16_t*)alloc((size_t)NBATCH * 1024 * 2);
    bf16_t* fb   = (bf16_t*)alloc((size_t)NBATCH * 1024 * 2);

    // shuffle + target
    shuffle_kernel<<<NBATCH, 256, 0, stream>>>(x, nz, img, out_target);

    // weight conversions (all to n-major bf16)
    convert_conv_w<<<(32 * 32 + 255) / 256, 256, 0, stream>>>(c1w, w1, 3, 32, 32);
    convert_conv_w<<<(288 * 64 + 255) / 256, 256, 0, stream>>>(c2w, w2, 32, 64, 288);
    convert_conv_w<<<(576 * 128 + 255) / 256, 256, 0, stream>>>(c3w, w3, 64, 128, 576);
    convert_fc1_w<<<(2048 * 1024) / 256, 256, 0, stream>>>(f1w, fw1);
    convert_fc_w<<<(1024 * 1024) / 256, 256, 0, stream>>>(f2w, fw2, 1024, 1024);
    convert_fc_w<<<(1024 * 1024) / 256, 256, 0, stream>>>(f3w, fw3, 1024, 1024);
    convert_fc_w<<<(1024 * 1024) / 256, 256, 0, stream>>>(f4w, fw4, 1024, 1024);
    convert_fc_w<<<(1024 * 256) / 256, 256, 0, stream>>>(f5w, fw5, 1024, 256);

    // conv stages, chunked over batch to bound workspace
    for (int c = 0; c < NBATCH / CHUNK; ++c) {
        const bf16_t* imgc = img + (size_t)c * CHUNK * 1024 * 3;
        bf16_t* act3c = act3 + (size_t)c * CHUNK * 16 * 128;
        conv_pool_gemm<3, 32, 32, 32, 32>
            <<<dim3(CHUNK * 1024 / 128, 1), 256, 0, stream>>>(imgc, w1, c1b, act1);
        conv_pool_gemm<32, 64, 16, 16, 64>
            <<<dim3(CHUNK * 256 / 128, 1), 256, 0, stream>>>(act1, w2, c2b, act2);
        conv_pool_gemm<64, 128, 8, 8, 64>
            <<<dim3(CHUNK * 64 / 128, 2), 256, 0, stream>>>(act2, w3, c3b, act3c);
    }

    // FC stack
    fc_gemm<true, true><<<dim3(NBATCH / 128, 1024 / 64), 256, 0, stream>>>(
        act3, fw1, f1b, fa, NBATCH, 2048, 1024);
    fc_gemm<true, true><<<dim3(NBATCH / 128, 1024 / 64), 256, 0, stream>>>(
        fa, fw2, f2b, fb, NBATCH, 1024, 1024);
    fc_gemm<true, true><<<dim3(NBATCH / 128, 1024 / 64), 256, 0, stream>>>(
        fb, fw3, f3b, fa, NBATCH, 1024, 1024);
    fc_gemm<true, true><<<dim3(NBATCH / 128, 1024 / 64), 256, 0, stream>>>(
        fa, fw4, f4b, fb, NBATCH, 1024, 1024);
    fc_gemm<false, false><<<dim3(NBATCH / 128, 256 / 64), 256, 0, stream>>>(
        fb, fw5, f5b, (void*)out_logits, NBATCH, 1024, 256);
}
